// VJEPA2RopeAttention_5282809774124
// MI455X (gfx1250) — matrix operand order repaired and checked by reading, not compiled, not run on hardware
//
#include <hip/hip_runtime.h>

typedef __attribute__((ext_vector_type(16))) _Float16 v16h;
typedef __attribute__((ext_vector_type(8)))  _Float16 v8h;
typedef __attribute__((ext_vector_type(4)))  _Float16 v4h;
typedef __attribute__((ext_vector_type(8)))  float    v8f;

#define WMMA_F16(a, b, c) \
  __builtin_amdgcn_wmma_f32_16x16x32_f16(false, (a), false, (b), (short)0, (c), false, false)

static __device__ __forceinline__ v16h join16(v8h lo, v8h hi) {
  return __builtin_shufflevector(lo, hi, 0,1,2,3,4,5,6,7,8,9,10,11,12,13,14,15);
}

struct Acc4 { v8f c00, c01, c10, c11; };

// 32x32 output tile of OUT = A[M,1024] * W[N,1024]^T per wave.
// 2 A-operands + 2 B-operands -> 4 WMMAs per k-step (register-level reuse).
static __device__ __forceinline__ Acc4 gemm32x32_k1024(
    const _Float16* __restrict__ A, const _Float16* __restrict__ W,
    int arow0, int wrow0, int lane) {
  const int klo = (lane < 16) ? 0 : 8;
  const int kb  = (lane < 16) ? 0 : 16;
  const _Float16* a0 = A + (size_t)arow0 * 1024;
  const _Float16* a1 = a0 + (size_t)16 * 1024;
  const _Float16* w0 = W + (size_t)wrow0 * 1024;
  const _Float16* w1 = w0 + (size_t)16 * 1024;
  Acc4 acc = {{}, {}, {}, {}};
#pragma unroll 2
  for (int kk = 0; kk < 1024; kk += 32) {
    v16h A0 = join16(*(const v8h*)(a0 + kk + klo), *(const v8h*)(a0 + kk + 16 + klo));
    v16h A1 = join16(*(const v8h*)(a1 + kk + klo), *(const v8h*)(a1 + kk + 16 + klo));
    v16h B0 = join16(*(const v8h*)(w0 + kk + klo), *(const v8h*)(w0 + kk + 16 + klo));
    v16h B1 = join16(*(const v8h*)(w1 + kk + klo), *(const v8h*)(w1 + kk + 16 + klo));
    acc.c00 = WMMA_F16(A0, B0, acc.c00);
    acc.c01 = WMMA_F16(A0, B1, acc.c01);
    acc.c10 = WMMA_F16(A1, B0, acc.c10);
    acc.c11 = WMMA_F16(A1, B1, acc.c11);
  }
  return acc;
}

// ---------------- kernel 0: f32 -> f16 conversion of X and 4 weight mats ----
__global__ __launch_bounds__(256) void cvt_kernel(
    const float* __restrict__ X, const float* __restrict__ Wq,
    const float* __restrict__ Wk, const float* __restrict__ Wv,
    const float* __restrict__ Wp, _Float16* __restrict__ Xh,
    _Float16* __restrict__ Wh) {
  const size_t gid = (size_t)blockIdx.x * blockDim.x + threadIdx.x;
  const size_t i = gid * 4;
  const float* src;
  _Float16* dst;
  if (i < (size_t)4194304) {            // X: 4096*1024 floats
    src = X + i;  dst = Xh + i;
  } else {                               // weights: 4 * 1024*1024 floats
    const size_t o = i - 4194304;
    const int mat = (int)(o >> 20);
    const size_t w = o & 1048575;
    src = (mat == 0) ? (Wq + w) : (mat == 1) ? (Wk + w) : (mat == 2) ? (Wv + w) : (Wp + w);
    dst = Wh + o;
  }
  const float4 f = *(const float4*)src;
  v4h hv;
  hv[0] = (_Float16)f.x; hv[1] = (_Float16)f.y;
  hv[2] = (_Float16)f.z; hv[3] = (_Float16)f.w;
  *(v4h*)dst = hv;
}

// ---------------- QKV epilogue: bias + fused 3D RoPE + layout stores --------
static __device__ __forceinline__ void qkv_epilogue(
    v8f c, int mat, int rowbase, int j, float bj, int lane,
    _Float16* __restrict__ Qh, _Float16* __restrict__ Kh,
    _Float16* __restrict__ Vt) {
  const int mbase = (lane < 16) ? 0 : 8;
#pragma unroll
  for (int r = 0; r < 8; ++r) c[r] += bj;
  const int h = j >> 6, d = j & 63;
  if (mat < 2) {
    // 3 RoPE segments of 20 dims; pair partner = feature j^1 = lane^1.
    const int seg = d / 20;                     // 3 => passthrough
    const int dl = d - seg * 20;
    const float omega = __expf(-0.9210340372f * (float)(dl % 10)); // 10000^{-(dl%10)/10}
    const float sgn = (d & 1) ? 1.0f : -1.0f;
#pragma unroll
    for (int r = 0; r < 8; ++r) {
      const float x = c[r];
      const float px = __shfl_xor(x, 1, 32);    // executed by all lanes
      const int npos = (rowbase + mbase + r) & 2047;
      const int pos = (seg == 0) ? (npos >> 8)
                    : (seg == 1) ? ((npos >> 4) & 15) : (npos & 15);
      const float ang = (float)pos * omega;
      const float res = x * __cosf(ang) + sgn * px * __sinf(ang);
      c[r] = (d < 60) ? res : x;
    }
    _Float16* dst = (mat == 0) ? Qh : Kh;
#pragma unroll
    for (int r = 0; r < 8; ++r)
      dst[(size_t)(rowbase + mbase + r) * 1024 + j] = (_Float16)c[r];
  } else {
    // V transposed [B,H,D,N]; 8 consecutive rows -> one 16B store
    const int i0 = rowbase + mbase;
    const int bb = i0 >> 11;
    const int npos0 = i0 & 2047;
    v8h vv;
#pragma unroll
    for (int r = 0; r < 8; ++r) vv[r] = (_Float16)c[r];
    *(v8h*)(Vt + (size_t)((bb * 16 + h) * 64 + d) * 2048 + npos0) = vv;
  }
}

// ---------------- kernel 1: QKV projection (32x32 tile / wave) --------------
__global__ __launch_bounds__(128) void qkv_kernel(
    const _Float16* __restrict__ Xh, const _Float16* __restrict__ Wh,
    const float* __restrict__ bq, const float* __restrict__ bk,
    const float* __restrict__ bv, _Float16* __restrict__ Qh,
    _Float16* __restrict__ Kh, _Float16* __restrict__ Vt) {
  const int lane = threadIdx.x & 31;
  const int wave = blockIdx.x * 4 + (threadIdx.x >> 5);
  const int mat = wave >> 12;                 // 4096 waves per matrix
  const int rem = wave & 4095;
  const int rowtile = rem >> 5, coltile = rem & 31;
  const int n = lane & 15;

  Acc4 acc = gemm32x32_k1024(Xh, Wh + (size_t)mat * 1048576,
                             rowtile * 32 + n, coltile * 32 + n, lane);

  const float* bias = (mat == 0) ? bq : (mat == 1) ? bk : bv;
  const int j0 = coltile * 32 + n, j1 = j0 + 16;
  const int r0 = rowtile * 32, r1 = r0 + 16;
  const float bj0 = bias[j0], bj1 = bias[j1];
  qkv_epilogue(acc.c00, mat, r0, j0, bj0, lane, Qh, Kh, Vt);
  qkv_epilogue(acc.c01, mat, r0, j1, bj1, lane, Qh, Kh, Vt);
  qkv_epilogue(acc.c10, mat, r1, j0, bj0, lane, Qh, Kh, Vt);
  qkv_epilogue(acc.c11, mat, r1, j1, bj1, lane, Qh, Kh, Vt);
}

// ---------------- kernel 2: flash attention --------------------------------
// 4 waves / block share one (b,h); K/V chunks staged in LDS with async loads
// (double-buffered). S^T = K*Q^T so score C-layout == P's A-operand layout.
#define LDS_KSTRIDE 72          // halves per K row (64 + 8 pad, conflict-free)
#define LDS_VSTRIDE 40          // halves per V row (32 + 8 pad)
#define LDS_VBASE   2304        // halves: 32 * 72
#define LDS_BUFSZ   4864        // halves: 2304 + 64*40

__global__ __launch_bounds__(128) void attn_kernel(
    const _Float16* __restrict__ Qh, const _Float16* __restrict__ Kh,
    const _Float16* __restrict__ Vt, _Float16* __restrict__ Ch) {
  __shared__ _Float16 sbuf[2][LDS_BUFSZ];
  const int t = threadIdx.x;
  const int lane = t & 31;
  const int w = t >> 5;
  const int bh = blockIdx.x >> 5;             // 0..31 (b*16+h)
  const int qtile = (blockIdx.x & 31) * 4 + w;
  const int b = bh >> 4, h = bh & 15;
  const int n = lane & 15;
  const int klo = (lane < 16) ? 0 : 8;
  const int kb  = (lane < 16) ? 0 : 16;
  const int obase = klo;

  const _Float16* Kb = Kh + (size_t)(b * 2048) * 1024 + h * 64;
  const _Float16* Vb = Vt + (size_t)(bh * 64) * 2048;

  // Q as B-operand, held in registers for the whole loop
  const _Float16* qrow = Qh + (size_t)(b * 2048 + qtile * 16 + n) * 1024 + h * 64;
  const v16h qb0 = join16(*(const v8h*)(qrow + klo), *(const v8h*)(qrow + 16 + klo));
  const v16h qb1 = join16(*(const v8h*)(qrow + 32 + klo), *(const v8h*)(qrow + 48 + klo));

  // cooperative async stage of one 32-key chunk of K and V into sbuf[bb]
  auto issue = [&](int kc, int bb) {
    const unsigned base = (unsigned)(size_t)(&sbuf[bb][0]); // LDS byte offset
#pragma unroll
    for (int uu = 0; uu < 2; ++uu) {              // K: 32 rows x 128B
      const int u = t + uu * 128;                 // 256 x 16B units
      const int key = u >> 3, dblk = u & 7;
      const _Float16* g = Kb + (size_t)(kc + key) * 1024 + dblk * 8;
      const unsigned l = base + (unsigned)(key * (LDS_KSTRIDE * 2) + dblk * 16);
      asm volatile("global_load_async_to_lds_b128 %0, %1, off"
                   :: "v"(l), "v"(g) : "memory");
    }
#pragma unroll
    for (int uu = 0; uu < 2; ++uu) {              // V: 64 rows x 64B
      const int u = t + uu * 128;                 // 256 x 16B units
      const int d = u >> 2, kblk = u & 3;
      const _Float16* g = Vb + (size_t)d * 2048 + kc + kblk * 8;
      const unsigned l = base + (unsigned)(LDS_VBASE * 2 + d * (LDS_VSTRIDE * 2) + kblk * 16);
      asm volatile("global_load_async_to_lds_b128 %0, %1, off"
                   :: "v"(l), "v"(g) : "memory");
    }
  };

  v8f o0 = {}, o1 = {}, o2 = {}, o3 = {};
  float m_run = -1e30f, l_run = 0.0f;

  issue(0, 0);
  for (int i = 0; i < 64; ++i) {
    asm volatile("s_wait_asynccnt 0x0" ::: "memory");  // my chunk-i loads done
    __syncthreads();                                    // everyone's done; prev buf free
    if (i + 1 < 64) issue((i + 1) * 32, (i + 1) & 1);
    const _Float16* sk = &sbuf[i & 1][0];
    const _Float16* sv = &sbuf[i & 1][LDS_VBASE];

    // ---- S^T tiles: keys 0..15 and 16..31 of the chunk (A from LDS) ----
    v8f c0 = {}, c1 = {};
    {
      const _Float16* kr = sk + (size_t)n * LDS_KSTRIDE;
      v16h a  = join16(*(const v8h*)(kr + klo),      *(const v8h*)(kr + 16 + klo));
      v16h a2 = join16(*(const v8h*)(kr + 32 + klo), *(const v8h*)(kr + 48 + klo));
      c0 = WMMA_F16(a, qb0, c0);
      c0 = WMMA_F16(a2, qb1, c0);
    }
    {
      const _Float16* kr = sk + (size_t)(16 + n) * LDS_KSTRIDE;
      v16h a  = join16(*(const v8h*)(kr + klo),      *(const v8h*)(kr + 16 + klo));
      v16h a2 = join16(*(const v8h*)(kr + 32 + klo), *(const v8h*)(kr + 48 + klo));
      c1 = WMMA_F16(a, qb0, c1);
      c1 = WMMA_F16(a2, qb1, c1);
    }
    // ---- online softmax (row q lives in lanes q and q+16) ----
    float mloc = -1e30f;
#pragma unroll
    for (int r = 0; r < 8; ++r) {
      c0[r] *= 0.125f; c1[r] *= 0.125f;       // D^-0.5
      mloc = fmaxf(mloc, fmaxf(c0[r], c1[r]));
    }
    mloc = fmaxf(mloc, __shfl_xor(mloc, 16, 32));
    const float m_new = fmaxf(m_run, mloc);
    const float alpha = __expf(m_run - m_new);
    float sloc = 0.0f;
#pragma unroll
    for (int r = 0; r < 8; ++r) {
      c0[r] = __expf(c0[r] - m_new);
      c1[r] = __expf(c1[r] - m_new);
      sloc += c0[r] + c1[r];
    }
    sloc += __shfl_xor(sloc, 16, 32);
    l_run = l_run * alpha + sloc;
    m_run = m_new;
#pragma unroll
    for (int r = 0; r < 8; ++r) {
      const float ar = __shfl(alpha, obase + r, 32);
      o0[r] *= ar; o1[r] *= ar; o2[r] *= ar; o3[r] *= ar;
    }
    // ---- P as A-operand (direct layout match with S^T C-layout) ----
    v16h pa;
#pragma unroll
    for (int r = 0; r < 8; ++r) {
      pa[r]     = (_Float16)c0[r];            // keys klo..klo+7
      pa[r + 8] = (_Float16)c1[r];            // keys 16+klo..
    }
    // ---- P*V: V B-operand from LDS (row = d, contiguous keys) ----
#pragma unroll
    for (int dt = 0; dt < 4; ++dt) {
      const _Float16* vr = sv + (size_t)(dt * 16 + n) * LDS_VSTRIDE + klo;
      v16h vb = join16(*(const v8h*)(vr), *(const v8h*)(vr + 16));
      if (dt == 0) o0 = WMMA_F16(pa, vb, o0);
      else if (dt == 1) o1 = WMMA_F16(pa, vb, o1);
      else if (dt == 2) o2 = WMMA_F16(pa, vb, o2);
      else o3 = WMMA_F16(pa, vb, o3);
    }
  }
  // ---- finalize: divide by l, store ctx f16 [B,N,H*D] ----
  const float linv = 1.0f / l_run;
  _Float16* crow = Ch + (size_t)(b * 2048 + qtile * 16 + obase) * 1024 + h * 64 + n;
#pragma unroll
  for (int r = 0; r < 8; ++r) {
    const float lr = __shfl(linv, obase + r, 32);
    crow[(size_t)r * 1024 +  0] = (_Float16)(o0[r] * lr);
    crow[(size_t)r * 1024 + 16] = (_Float16)(o1[r] * lr);
    crow[(size_t)r * 1024 + 32] = (_Float16)(o2[r] * lr);
    crow[(size_t)r * 1024 + 48] = (_Float16)(o3[r] * lr);
  }
}

// ---------------- kernel 3: output projection (32x32 tile / wave) ----------
__global__ __launch_bounds__(128) void proj_kernel(
    const _Float16* __restrict__ Ch, const _Float16* __restrict__ Wp,
    const float* __restrict__ bp, float* __restrict__ out) {
  const int lane = threadIdx.x & 31;
  const int wave = blockIdx.x * 4 + (threadIdx.x >> 5);
  const int rowtile = wave >> 5, coltile = wave & 31;
  const int n = lane & 15;
  const int mbase = (lane < 16) ? 0 : 8;

  Acc4 acc = gemm32x32_k1024(Ch, Wp, rowtile * 32 + n, coltile * 32 + n, lane);

  const int j0 = coltile * 32 + n, j1 = j0 + 16;
  const int r0 = rowtile * 32, r1 = r0 + 16;
  const float bj0 = bp[j0], bj1 = bp[j1];
#pragma unroll
  for (int r = 0; r < 8; ++r) {
    out[(size_t)(r0 + mbase + r) * 1024 + j0] = acc.c00[r] + bj0;
    out[(size_t)(r0 + mbase + r) * 1024 + j1] = acc.c01[r] + bj1;
    out[(size_t)(r1 + mbase + r) * 1024 + j0] = acc.c10[r] + bj0;
    out[(size_t)(r1 + mbase + r) * 1024 + j1] = acc.c11[r] + bj1;
  }
}

extern "C" void kernel_launch(void* const* d_in, const int* in_sizes, int n_in,
                              void* d_out, int out_size, void* d_ws, size_t ws_size,
                              hipStream_t stream) {
  const float* X  = (const float*)d_in[0];
  const float* Wq = (const float*)d_in[1];
  const float* bq = (const float*)d_in[2];
  const float* Wk = (const float*)d_in[3];
  const float* bk = (const float*)d_in[4];
  const float* Wv = (const float*)d_in[5];
  const float* bv = (const float*)d_in[6];
  const float* Wp = (const float*)d_in[7];
  const float* bp = (const float*)d_in[8];
  float* out = (float*)d_out;

  // workspace: 6 x 4Mi f16 buffers = 48 MiB
  const size_t M4 = 4194304;
  _Float16* Xh = (_Float16*)d_ws;
  _Float16* Wh = Xh + M4;       // [4][1024][1024] f16: Wq,Wk,Wv,Wp
  _Float16* Qh = Wh + M4;
  _Float16* Kh = Qh + M4;
  _Float16* Vt = Kh + M4;       // [B,H,D,N]
  _Float16* Ch = Vt + M4;       // ctx f16 [B,N,HS]

  cvt_kernel <<<8192, 256, 0, stream>>>(X, Wq, Wk, Wv, Wp, Xh, Wh);
  qkv_kernel <<<3072, 128, 0, stream>>>(Xh, Wh, bq, bk, bv, Qh, Kh, Vt);
  attn_kernel<<<1024, 128, 0, stream>>>(Qh, Kh, Vt, Ch);
  proj_kernel<<<1024, 128, 0, stream>>>(Ch, Wh + (size_t)3 * 1048576, bp, out);
}
